// ATConv2d_58162447122907
// MI455X (gfx1250) — matrix operand order repaired and checked by reading, not compiled, not run on hardware
//
#include <hip/hip_runtime.h>
#include <hip/hip_bf16.h>
#include <math.h>

typedef _Float16 v16h __attribute__((ext_vector_type(16)));
typedef _Float16 v8h  __attribute__((ext_vector_type(8)));
typedef float    v8f  __attribute__((ext_vector_type(8)));

#define NB    8
#define CIN   64
#define COUT  64
#define HH    128
#define WW    128
#define CTXC  16
#define FEAT  576          // CIN * 9, reordered as (tap, c)
#define HWSZ  16384
#define KSCALE     4096.0f
#define INV_KSCALE (1.0f / 4096.0f)
#define CSTRIDE 72         // padded channel stride (halves) -> 144B rows, 16B aligned

// ---------------- kernel 1: mean over H,W per (b,c) ----------------
__global__ void atc_mean_kernel(const float* __restrict__ x, float* __restrict__ mean_x) {
    int bc = blockIdx.x;                      // 0..511
    const float* p = x + (size_t)bc * HWSZ;
    float s = 0.f;
    for (int i = threadIdx.x; i < HWSZ; i += 256) s += p[i];
    __shared__ float red[256];
    red[threadIdx.x] = s;
    __syncthreads();
    for (int off = 128; off > 0; off >>= 1) {
        if (threadIdx.x < off) red[threadIdx.x] += red[threadIdx.x + off];
        __syncthreads();
    }
    if (threadIdx.x == 0) mean_x[bc] = red[0] * (1.0f / HWSZ);
}

// ---------------- kernel 2: ctx_vec = ctx_w @ mean_x + ctx_b ----------------
__global__ void atc_ctx_kernel(const float* __restrict__ mean_x,
                               const float* __restrict__ ctx_w,
                               const float* __restrict__ ctx_b,
                               float* __restrict__ ctx_vec) {
    int t = threadIdx.x;
    if (t < NB * CTXC) {
        int b = t / CTXC, o = t % CTXC;
        float s = ctx_b[o];
        for (int c = 0; c < CIN; ++c) s += ctx_w[o * CIN + c] * mean_x[b * CIN + c];
        ctx_vec[t] = s;
    }
}

// ---- kernel 3: generate kernels, tanh, mean-center, fold value_w, pack f16 ----
// Output Kp[b*COUT + o][tap*64 + c] = SCALE * sum_c2 (tanh(kg(b,o,c2,tap)) - lam*mean) * value_w[c2][c]
__global__ void atc_kgen_kernel(const float* __restrict__ ctx_vec,
                                const float* __restrict__ kg_w,
                                const float* __restrict__ kg_b,
                                const float* __restrict__ gamma,
                                const float* __restrict__ value_w,
                                _Float16* __restrict__ Kp) {
    int blk = blockIdx.x;                     // b*COUT + o
    int b = blk >> 6, o = blk & 63;
    int t = threadIdx.x;
    __shared__ float kern_s[FEAT];            // raw kernels, original order (c2*9 + tap)
    __shared__ float red[256];
    __shared__ float ctx_s[CTXC];
    if (t < CTXC) ctx_s[t] = ctx_vec[b * CTXC + t];
    __syncthreads();

    float part = 0.f;
    for (int e = t; e < FEAT; e += 256) {
        size_t row = (size_t)o * FEAT + e;
        const float* wrow = kg_w + row * CTXC;
        float v = kg_b[row];
        #pragma unroll
        for (int j = 0; j < CTXC; ++j) v += ctx_s[j] * wrow[j];
        v = tanhf(v);
        kern_s[e] = v;
        part += v;
    }
    red[t] = part;
    __syncthreads();
    for (int off = 128; off > 0; off >>= 1) {
        if (t < off) red[t] += red[t + off];
        __syncthreads();
    }
    float lam = 1.0f / (1.0f + expf(-gamma[o]));
    float sub = lam * red[0] * (1.0f / FEAT);
    __syncthreads();

    // fold value projection; reorder K dim to (tap, c); scale and pack to f16
    for (int e = t; e < FEAT; e += 256) {
        int tap = e >> 6, c = e & 63;
        float acc = 0.f;
        for (int c2 = 0; c2 < CIN; ++c2)
            acc += (kern_s[c2 * 9 + tap] - sub) * value_w[c2 * CIN + c];
        Kp[(size_t)blk * FEAT + tap * CIN + c] = (_Float16)(acc * KSCALE);
    }
}

// ---------------- kernel 4: WMMA dynamic conv ----------------
// out[b,o,h,w] = relu( (1/SCALE) * sum_{tap,c} Kp[b,o,tap,c] * x[b,c,h+kh-1,w+kw-1] + bias[o] )
__global__ void __launch_bounds__(256, 2)
atc_conv_kernel(const float* __restrict__ x,
                const _Float16* __restrict__ Kp,
                const float* __restrict__ bias,
                float* __restrict__ out) {
    __shared__ __align__(16) _Float16 xs[3 * 34 * CSTRIDE];  // [kh][col][c]

    int b     = blockIdx.y;
    int tile  = blockIdx.x;            // 0..511 : h * 4 + wtile
    int h     = tile >> 2;
    int wbase = (tile & 3) * 32;

    int t    = threadIdx.x;
    int wave = t >> 5, lane = t & 31;

    // ---- stage x tile (64 ch x 3 rows x 34 cols) into LDS as f16 ----
    for (int it = 0; it < 24; ++it) {
        int pair = it * 8 + wave;      // 0..191 = c*3 + kh
        int c  = pair / 3;
        int kh = pair % 3;
        int hr = h + kh - 1;
        bool rowok = (hr >= 0) && (hr < HH);
        const float* xrow = x + (((size_t)(b * CIN + c)) * HH + hr) * WW;
        {
            int col = lane;
            int wc  = wbase - 1 + col;
            float v = (rowok && wc >= 0 && wc < WW) ? xrow[wc] : 0.f;
            xs[(kh * 34 + col) * CSTRIDE + c] = (_Float16)v;
        }
        if (lane < 2) {
            int col = 32 + lane;
            int wc  = wbase - 1 + col;
            float v = (rowok && wc < WW) ? xrow[wc] : 0.f;
            xs[(kh * 34 + col) * CSTRIDE + c] = (_Float16)v;
        }
    }
    __syncthreads();

    int mtile  = wave & 3;             // M tile (output channels)
    int ntile  = wave >> 2;            // N tile (pixels)
    int nlocal = lane & 15;
    int laneHi = lane >> 4;
    int px     = ntile * 16 + nlocal;  // pixel col within 32-wide strip

    v8f acc = {0.f, 0.f, 0.f, 0.f, 0.f, 0.f, 0.f, 0.f};

    // A row for this lane: M = mtile*16 + (lane&15)
    const _Float16* Ab = Kp + ((size_t)(b * COUT) + mtile * 16 + nlocal) * FEAT;

    #pragma unroll
    for (int tap = 0; tap < 9; ++tap) {
        int kh = tap / 3, kw = tap % 3;
        #pragma unroll
        for (int ch = 0; ch < 2; ++ch) {
            int kbase = tap * 64 + ch * 32;
            // A fragment (16x32 f16): elems 0..15 -> K = laneHi*16 + [0,16)
            v8h a0 = *reinterpret_cast<const v8h*>(Ab + kbase + laneHi * 16);
            v8h a1 = *reinterpret_cast<const v8h*>(Ab + kbase + laneHi * 16 + 8);
            v16h afrag = __builtin_shufflevector(a0, a1,
                0, 1, 2, 3, 4, 5, 6, 7, 8, 9, 10, 11, 12, 13, 14, 15);
            // B fragment (32x16 f16): lane holds column N = lane&15,
            //                         K = ch*32 + laneHi*16 + [0,16)
            const _Float16* bp =
                &xs[(kh * 34 + px + kw) * CSTRIDE + ch * 32 + laneHi * 16];
            v8h b0 = *reinterpret_cast<const v8h*>(bp);
            v8h b1 = *reinterpret_cast<const v8h*>(bp + 8);
            v16h bfrag = __builtin_shufflevector(b0, b1,
                0, 1, 2, 3, 4, 5, 6, 7, 8, 9, 10, 11, 12, 13, 14, 15);

            acc = __builtin_amdgcn_wmma_f32_16x16x32_f16(
                false, afrag, false, bfrag, (short)0, acc, false, false);
        }
    }

    // ---- epilogue: unscale, bias, relu, store ----
    int wcol = wbase + px;
    #pragma unroll
    for (int r = 0; r < 8; ++r) {
        int o = mtile * 16 + r + laneHi * 8;   // C/D layout: VGPR r -> M=r (lanes<16), M=r+8 (lanes>=16)
        float vo = acc[r] * INV_KSCALE + bias[o];
        vo = vo > 0.f ? vo : 0.f;
        out[(((size_t)(b * COUT + o)) * HH + h) * WW + wcol] = vo;
    }
}

// ---------------- launcher ----------------
extern "C" void kernel_launch(void* const* d_in, const int* in_sizes, int n_in,
                              void* d_out, int out_size, void* d_ws, size_t ws_size,
                              hipStream_t stream) {
    const float* x       = (const float*)d_in[0];
    const float* ctx_w   = (const float*)d_in[1];
    const float* ctx_b   = (const float*)d_in[2];
    const float* kg_w    = (const float*)d_in[3];
    const float* kg_b    = (const float*)d_in[4];
    const float* gamma   = (const float*)d_in[5];
    const float* bias    = (const float*)d_in[6];
    const float* value_w = (const float*)d_in[7];
    float* out = (float*)d_out;

    char* ws = (char*)d_ws;
    float*    mean_x  = (float*)ws;              // 512 f32   @ 0
    float*    ctx_vec = (float*)(ws + 2048);     // 128 f32   @ 2048
    _Float16* Kp      = (_Float16*)(ws + 4096);  // 294912 f16 @ 4096 (~576 KB)

    atc_mean_kernel<<<NB * CIN, 256, 0, stream>>>(x, mean_x);
    atc_ctx_kernel<<<1, 128, 0, stream>>>(mean_x, ctx_w, ctx_b, ctx_vec);
    atc_kgen_kernel<<<NB * COUT, 256, 0, stream>>>(ctx_vec, kg_w, kg_b, gamma, value_w, Kp);

    dim3 grid(512, NB);
    atc_conv_kernel<<<grid, 256, 0, stream>>>(x, Kp, bias, out);
}
